// ScoreNet_embedding_4544075399863
// MI455X (gfx1250) — compile-verified, matrix operand order checked by reading, not run on hardware
//
#include <hip/hip_runtime.h>
#include <hip/hip_bf16.h>
#include <math.h>

// ---------------------------------------------------------------------------
// ScoreNet fused kernel for MI455X (gfx1250, wave32, WMMA).
// B=8192 samples, N=100 points, 12 moments, MLP 12->64->128->12.
// MLP matmuls: v_wmma_f32_16x16x32_f16, 16 batch rows per wave tile.
// Pointwise phases are VALU-bound -> branchless fast-erf GELU, fused
// centering+moments, float2/float4 data movement, Horner jacobian.
// ---------------------------------------------------------------------------

typedef _Float16 v16h __attribute__((ext_vector_type(16)));
typedef float    v8f  __attribute__((ext_vector_type(8)));

#define NBATCH   8192
#define NPTS     100
#define NMOM     12
#define HDIM     64
#define ODIM     128
#define WAVES_PB 2           // waves per block
#define FRAG_HALVES 512      // 32 lanes * 16 halves per fragment tile
#define N_TILES  25          // 4 (W1) + 16 (W2a) + 4 (W2b) + 1 (W2c)

// Branchless GELU: 0.5*x*(1+erf(x/sqrt(2))), erf via A&S 7.1.26 (|err|<1.5e-7).
// rcp + exp (TRANS) + ~10 FMA, no EXEC divergence around WMMA code.
__device__ __forceinline__ float gelu_fast(float x) {
    float z  = x * 0.70710678118654752f;
    float az = fabsf(z);
    float t  = __builtin_amdgcn_rcpf(fmaf(0.3275911f, az, 1.0f));
    float p  = fmaf(t, fmaf(t, fmaf(t, fmaf(t, 1.061405429f, -1.453152027f),
                                    1.421413741f), -0.284496736f), 0.254829592f);
    float e  = __expf(-az * az);
    float erfa = fmaf(-p * t, e, 1.0f);          // erf(|z|)
    float erfz = copysignf(erfa, z);
    return 0.5f * x * (1.0f + erfz);
}

// K index inside a 16-wide f16 A/B fragment, per CDNA5 ISA 7.12.2:
// lane half (lane>=16) owns K offsets 8..15 / 24..31; element e: VGPR e/2, half e%2.
__device__ __forceinline__ int frag_k(int e, int kb) {
    return ((e >> 3) << 4) + kb + 2 * ((e >> 1) & 3) + (e & 1);
}

// ---------------------------------------------------------------------------
// Kernel 1: repack fp32 weights into f16 WMMA B-fragments in workspace.
// Tile map: [0..3]=W1(12x64,nt)  [4..19]=W2a(64x128, kc*8+nt)
//           [20..23]=W2b(128x12, kc)  [24]=W2c(12x12)
// ---------------------------------------------------------------------------
__global__ void prep_frags(const float* __restrict__ W1,
                           const float* __restrict__ W2a,
                           const float* __restrict__ W2b,
                           const float* __restrict__ W2c,
                           _Float16* __restrict__ frag) {
    int tile = blockIdx.x;      // 25 tiles
    int lane = threadIdx.x;     // 32 lanes
    const float* W; int Kdim, fout, k0, n0;
    if (tile < 4)        { W = W1;  Kdim = 12;  fout = 64;  k0 = 0;                 n0 = tile * 16; }
    else if (tile < 20)  { int t2 = tile - 4;
                           W = W2a; Kdim = 64;  fout = 128; k0 = (t2 >> 3) * 32;    n0 = (t2 & 7) * 16; }
    else if (tile < 24)  { W = W2b; Kdim = 128; fout = 12;  k0 = (tile - 20) * 32;  n0 = 0; }
    else                 { W = W2c; Kdim = 12;  fout = 12;  k0 = 0;                 n0 = 0; }

    int col = n0 + (lane & 15);
    int kb  = (lane >> 4) * 8;
    _Float16* dst = frag + tile * FRAG_HALVES + lane * 16;
#pragma unroll
    for (int e = 0; e < 16; ++e) {
        int k = k0 + frag_k(e, kb);
        float v = (k < Kdim && col < fout) ? W[k * fout + col] : 0.0f;
        dst[e] = (_Float16)v;
    }
}

// ---------------------------------------------------------------------------
// Kernel 2: Gaussian-Fourier time embedding -> emb[64] = [sin,cos] @ embed_W + b
// ---------------------------------------------------------------------------
__global__ void prep_emb(const float* __restrict__ t,
                         const float* __restrict__ Wf,
                         const float* __restrict__ eW,
                         const float* __restrict__ eb,
                         float* __restrict__ emb) {
    int j = threadIdx.x;        // 64 threads
    float tt = t[0];
    float acc = eb[j];
#pragma unroll 4
    for (int i = 0; i < 32; ++i) {
        float xp = tt * Wf[i] * 6.28318530717958648f;
        acc += __sinf(xp) * eW[i * HDIM + j];
        acc += __cosf(xp) * eW[(32 + i) * HDIM + j];
    }
    emb[j] = acc;
}

// ---------------------------------------------------------------------------
// Kernel 3: fused moments -> WMMA MLP -> jacobian contraction.
// One wave (32 lanes) handles 16 batch samples. 2 waves per block.
// ---------------------------------------------------------------------------
__global__ __launch_bounds__(32 * WAVES_PB) void
score_main(const float* __restrict__ x, const float* __restrict__ t,
           const float* __restrict__ b1, const float* __restrict__ b2a,
           const float* __restrict__ b2b, const float* __restrict__ b2c,
           const _Float16* __restrict__ frag, const float* __restrict__ emb_g,
           float* __restrict__ out) {
    __shared__ __align__(16) float s_xc [WAVES_PB][16 * NPTS * 2];     // centered pts / u
    __shared__ __align__(16) float s_h  [WAVES_PB][16 * (HDIM + ODIM)];// h1 then h2
    __shared__ __align__(16) float s_hm [WAVES_PB][16 * NMOM];         // moments/h3/h
    __shared__ __align__(16) float s_mu [WAVES_PB][32];                // per-sample means
    __shared__ __align__(16) float s_emb[HDIM];

    const int w    = threadIdx.x >> 5;
    const int lane = threadIdx.x & 31;
    const int tileIdx = blockIdx.x * WAVES_PB + w;
    const int samp0   = tileIdx * 16;

    float* XC = s_xc[w];
    float* H1 = s_h[w];
    float* H2 = s_h[w] + 16 * HDIM;
    float* HM = s_hm[w];
    float* MU = s_mu[w];

    if (threadIdx.x < HDIM) s_emb[threadIdx.x] = emb_g[threadIdx.x];

    // ---- load x tile (16 samples x 200 floats) into LDS, float4 ----
    const float4* xg4 = (const float4*)(x + (size_t)samp0 * (NPTS * 2));
    float4* XC4 = (float4*)XC;
    __builtin_prefetch(xg4 + (16 * NPTS * 2) / 4, 0, 1);   // next tile
#pragma unroll 5
    for (int i = lane; i < (16 * NPTS * 2) / 4; i += 32) XC4[i] = xg4[i];
    __syncthreads();

    const int s    = lane & 15;
    const int half = lane >> 4;

    // ---- per-sample means: 2 lanes/sample, float2 loads, shfl reduce ----
    {
        const float2* xs2 = (const float2*)(XC + s * (2 * NPTS));
        float m0 = 0.f, m1 = 0.f;
        for (int n = half * 50; n < half * 50 + 50; ++n) {
            float2 v = xs2[n]; m0 += v.x; m1 += v.y;
        }
        m0 += __shfl_xor(m0, 16);
        m1 += __shfl_xor(m1, 16);
        if (half == 0) { MU[s * 2] = m0 * 0.01f; MU[s * 2 + 1] = m1 * 0.01f; }
    }
    __syncthreads();

    // ---- fused centering + 12 central moments (center written back) ----
    {
        float mu0 = MU[s * 2], mu1 = MU[s * 2 + 1];
        float mm[NMOM];
#pragma unroll
        for (int k = 0; k < NMOM; ++k) mm[k] = 0.f;
        float2* xs2 = (float2*)(XC + s * (2 * NPTS));
        for (int n = half * 50; n < half * 50 + 50; ++n) {
            float2 v = xs2[n];
            float a = v.x - mu0, b = v.y - mu1;
            xs2[n] = make_float2(a, b);
            float a2 = a * a, b2 = b * b;
            float a3 = a2 * a, b3 = b2 * b;
            mm[0]  = fmaf(a,  b,  mm[0]);   mm[1]  = fmaf(a,  a,  mm[1]);
            mm[2]  = fmaf(b,  b,  mm[2]);   mm[3]  = fmaf(a2, b,  mm[3]);
            mm[4]  = fmaf(a,  b2, mm[4]);   mm[5]  = fmaf(a2, a,  mm[5]);
            mm[6]  = fmaf(b2, b,  mm[6]);   mm[7]  = fmaf(a2, b2, mm[7]);
            mm[8]  = fmaf(a3, b,  mm[8]);   mm[9]  = fmaf(a,  b3, mm[9]);
            mm[10] = fmaf(a2, a2, mm[10]);  mm[11] = fmaf(b2, b2, mm[11]);
        }
#pragma unroll
        for (int k = 0; k < NMOM; ++k) mm[k] += __shfl_xor(mm[k], 16);
        if (half == 0) {
#pragma unroll
            for (int k = 0; k < NMOM; ++k) HM[s * NMOM + k] = mm[k];
        }
    }
    __syncthreads();

    const int row   = lane & 15;          // A-matrix row / D-matrix column mapping
    const int kb    = (lane >> 4) * 8;
    const int col16 = lane & 15;
    const int rhalf = (lane >> 4) * 8;

    // ---- layer 1: [16,12(->32)] @ [12,64], 4 WMMAs, bias in accumulator ----
    {
        v16h aF;
#pragma unroll
        for (int e = 0; e < 16; e += 2) {       // even K pairs -> float2 LDS loads
            int K = frag_k(e, kb);
            if (K < NMOM) {
                float2 v = *(const float2*)(&HM[row * NMOM + K]);
                aF[e] = (_Float16)v.x; aF[e + 1] = (_Float16)v.y;
            } else {
                aF[e] = (_Float16)0.0f; aF[e + 1] = (_Float16)0.0f;
            }
        }
#pragma unroll
        for (int nt = 0; nt < 4; ++nt) {
            int col = nt * 16 + col16;
            float bias = b1[col];
            v8f acc;
#pragma unroll
            for (int r = 0; r < 8; ++r) acc[r] = bias;  // bias same for all rows
            v16h bF = *(const v16h*)(frag + nt * FRAG_HALVES + lane * 16);
            acc = __builtin_amdgcn_wmma_f32_16x16x32_f16(false, aF, false, bF,
                                                         (short)0, acc, false, false);
#pragma unroll
            for (int r = 0; r < 8; ++r)
                H1[(rhalf + r) * HDIM + col] = gelu_fast(acc[r]);
        }
    }
    __syncthreads();

    // ---- layer 2a: [16,64] (+emb) @ [64,128], 2x8 WMMAs ----
    {
        v16h aK[2];
#pragma unroll
        for (int kc = 0; kc < 2; ++kc)
#pragma unroll
            for (int e = 0; e < 16; e += 2) {
                int kg = kc * 32 + frag_k(e, kb);
                float2 v  = *(const float2*)(&H1[row * HDIM + kg]);
                float2 em = *(const float2*)(&s_emb[kg]);
                aK[kc][e]     = (_Float16)(v.x + em.x);
                aK[kc][e + 1] = (_Float16)(v.y + em.y);
            }
#pragma unroll
        for (int nt = 0; nt < 8; ++nt) {
            int col = nt * 16 + col16;
            float bias = b2a[col];
            v8f acc;
#pragma unroll
            for (int r = 0; r < 8; ++r) acc[r] = bias;
#pragma unroll
            for (int kc = 0; kc < 2; ++kc) {
                v16h bF = *(const v16h*)(frag + (4 + kc * 8 + nt) * FRAG_HALVES + lane * 16);
                acc = __builtin_amdgcn_wmma_f32_16x16x32_f16(false, aK[kc], false, bF,
                                                             (short)0, acc, false, false);
            }
#pragma unroll
            for (int r = 0; r < 8; ++r)
                H2[(rhalf + r) * ODIM + col] = gelu_fast(acc[r]);
        }
    }
    __syncthreads();

    // ---- layer 2b: [16,128] @ [128,12(->16)], 4 WMMAs ----
    {
        float bias = b2b[col16 < NMOM ? col16 : 0];
        v8f acc;
#pragma unroll
        for (int r = 0; r < 8; ++r) acc[r] = bias;
#pragma unroll
        for (int kc = 0; kc < 4; ++kc) {
            v16h aF;
#pragma unroll
            for (int e = 0; e < 16; e += 2) {
                int kg = kc * 32 + frag_k(e, kb);
                float2 v = *(const float2*)(&H2[row * ODIM + kg]);
                aF[e] = (_Float16)v.x; aF[e + 1] = (_Float16)v.y;
            }
            v16h bF = *(const v16h*)(frag + (20 + kc) * FRAG_HALVES + lane * 16);
            acc = __builtin_amdgcn_wmma_f32_16x16x32_f16(false, aF, false, bF,
                                                         (short)0, acc, false, false);
        }
        __syncthreads();
        if (col16 < NMOM) {
#pragma unroll
            for (int r = 0; r < 8; ++r)
                HM[(rhalf + r) * NMOM + col16] = gelu_fast(acc[r]);
        }
    }
    __syncthreads();

    // ---- layer 2c: [16,12(->32)] @ [12,12(->16)], 1 WMMA; scale by 1/sqrt(t) ----
    {
        v16h aF;
#pragma unroll
        for (int e = 0; e < 16; e += 2) {
            int K = frag_k(e, kb);
            if (K < NMOM) {
                float2 v = *(const float2*)(&HM[row * NMOM + K]);
                aF[e] = (_Float16)v.x; aF[e + 1] = (_Float16)v.y;
            } else {
                aF[e] = (_Float16)0.0f; aF[e + 1] = (_Float16)0.0f;
            }
        }
        float bias = b2c[col16 < NMOM ? col16 : 0];
        v8f acc;
#pragma unroll
        for (int r = 0; r < 8; ++r) acc[r] = bias;
        v16h bF = *(const v16h*)(frag + 24 * FRAG_HALVES + lane * 16);
        acc = __builtin_amdgcn_wmma_f32_16x16x32_f16(false, aF, false, bF,
                                                     (short)0, acc, false, false);
        float sc = rsqrtf(t[0]);            // DIFF = 1.0
        __syncthreads();                    // all aF reads done before overwrite
        if (col16 < NMOM) {
#pragma unroll
            for (int r = 0; r < 8; ++r)
                HM[(rhalf + r) * NMOM + col16] = acc[r] * sc;
        }
    }
    __syncthreads();

    // ---- fused jacobian contraction:
    //   u_j(n) = sum_k h_k * d m_k / d xc_{n,j};  out = u - mean_n(u) ----
    {
        float hk[NMOM];
#pragma unroll
        for (int k = 0; k < NMOM; ++k) hk[k] = HM[s * NMOM + k];
        // pre-scaled coefficients
        float d1 = 2.f * hk[1], d2 = 2.f * hk[2], d3 = 2.f * hk[3];
        float d4 = 2.f * hk[4], d7 = 2.f * hk[7];
        float e5 = 3.f * hk[5], e6 = 3.f * hk[6], e8 = 3.f * hk[8], e9 = 3.f * hk[9];
        float f10 = 4.f * hk[10], f11 = 4.f * hk[11];

        float2* xs2 = (float2*)(XC + s * (2 * NPTS));
        float su0 = 0.f, su1 = 0.f;
        for (int n = half * 50; n < half * 50 + 50; ++n) {
            float2 v = xs2[n];
            float a = v.x, b = v.y;
            float a2 = a * a, b2 = b * b;
            // u0 = b*(hk0+hk4*b+hk9*b2) + a*(2hk1 + b*(2hk3+2hk7*b)) + a2*(3hk5+3hk8*b+4hk10*a)
            float pb = fmaf(hk[9], b2, fmaf(hk[4], b, hk[0]));
            float pa = fmaf(f10, a, fmaf(e8, b, e5));
            float pm = fmaf(d7, b, d3);
            float u0 = fmaf(b, pb, fmaf(a2, pa, a * fmaf(b, pm, d1)));
            // u1 = a*(hk0+hk3*a+hk8*a2) + b*(2hk2 + a*(2hk4+2hk7*a)) + b2*(3hk6+3hk9*a+4hk11*b)
            float qa = fmaf(hk[8], a2, fmaf(hk[3], a, hk[0]));
            float qb = fmaf(f11, b, fmaf(e9, a, e6));
            float qm = fmaf(d7, a, d4);
            float u1 = fmaf(a, qa, fmaf(b2, qb, b * fmaf(a, qm, d2)));
            xs2[n] = make_float2(u0, u1);    // reuse XC as u buffer
            su0 += u0; su1 += u1;
        }
        su0 += __shfl_xor(su0, 16);
        su1 += __shfl_xor(su1, 16);
        su0 *= (1.0f / NPTS);
        su1 *= (1.0f / NPTS);
        float2* og2 = (float2*)(out + (size_t)samp0 * (2 * NPTS) + s * (2 * NPTS));
        for (int n = half * 50; n < half * 50 + 50; ++n) {
            float2 u = xs2[n];
            og2[n] = make_float2(u.x - su0, u.y - su1);
        }
    }
}

// ---------------------------------------------------------------------------
extern "C" void kernel_launch(void* const* d_in, const int* in_sizes, int n_in,
                              void* d_out, int out_size, void* d_ws, size_t ws_size,
                              hipStream_t stream) {
    const float* x   = (const float*)d_in[0];
    const float* t   = (const float*)d_in[1];
    // d_in[2] = K exponent pairs: hardcoded in the kernels
    const float* Wf  = (const float*)d_in[3];
    const float* eW  = (const float*)d_in[4];
    const float* eb  = (const float*)d_in[5];
    const float* W1  = (const float*)d_in[6];
    const float* b1  = (const float*)d_in[7];
    const float* W2a = (const float*)d_in[8];
    const float* b2a = (const float*)d_in[9];
    const float* W2b = (const float*)d_in[10];
    const float* b2b = (const float*)d_in[11];
    const float* W2c = (const float*)d_in[12];
    const float* b2c = (const float*)d_in[13];
    float* out = (float*)d_out;

    _Float16* frag = (_Float16*)d_ws;                         // 25 KB of fragments
    float*    emb  = (float*)((char*)d_ws + N_TILES * FRAG_HALVES * sizeof(_Float16));

    prep_frags<<<N_TILES, 32, 0, stream>>>(W1, W2a, W2b, W2c, frag);
    prep_emb<<<1, HDIM, 0, stream>>>(t, Wf, eW, eb, emb);

    const int blocks = NBATCH / (16 * WAVES_PB);              // 256
    score_main<<<blocks, 32 * WAVES_PB, 0, stream>>>(x, t, b1, b2a, b2b, b2c,
                                                     frag, emb, out);
}
